// Attention_7112465842253
// MI455X (gfx1250) — matrix operand order repaired and checked by reading, not compiled, not run on hardware
//
#include <hip/hip_runtime.h>
#include <hip/hip_bf16.h>

// ---------------------------------------------------------------------------
// Fused attention block for MI455X (gfx1250, wave32, WMMA bf16 + TDM path)
//   qkv = x @ Wqkv + bqkv
//   q,k = RMSNorm(full 1024-ch) * scale ; RoPE (interleaved pairs, token0 skip)
//   attn = softmax(q k^T / 8) v      (16 heads, head_dim 64, full attention)
//   out  = attn @ Wout + bout
// ---------------------------------------------------------------------------

typedef unsigned short u16;
typedef __attribute__((ext_vector_type(16))) __bf16 v16bf;
typedef __attribute__((ext_vector_type(8)))  float  v8f;

union Frag16 { uint4 u4[2]; v16bf v; };   // 16 bf16 = 32 bytes = 8 VGPRs

__device__ __forceinline__ u16 f2bf(float f) {
    union { float f; unsigned u; } x; x.f = f;
    unsigned r = x.u + 0x7FFFu + ((x.u >> 16) & 1u);   // round-nearest-even
    return (u16)(r >> 16);
}

#ifndef __has_builtin
#define __has_builtin(x) 0
#endif
#if __has_builtin(__builtin_amdgcn_tensor_load_to_lds) && \
    __has_builtin(__builtin_amdgcn_s_wait_tensorcnt)
#define USE_TDM 1
#else
#define USE_TDM 0
#endif

#if USE_TDM
typedef __attribute__((ext_vector_type(4))) unsigned int v4u;
typedef __attribute__((ext_vector_type(8))) int          v8i;
typedef __attribute__((ext_vector_type(4))) int          v4i;

// 2-D tile load via Tensor Data Mover (D# per CDNA5 ISA)
// gsrc points at the tile start inside a (th x tw) bf16 tensor of row stride
// `stride` elements; tile is (tile_h rows x tile_w elems), packed to LDS.
__device__ __forceinline__ void tdm_load_2d(const u16* gsrc, unsigned lds_off,
                                            unsigned tw, unsigned th,
                                            unsigned stride,
                                            unsigned tile_w, unsigned tile_h) {
    unsigned long long ga = (unsigned long long)gsrc;
    v4u g0;
    g0[0] = 1u;                                                 // count=1, user
    g0[1] = lds_off;                                            // lds_addr
    g0[2] = (unsigned)ga;                                       // global[31:0]
    g0[3] = (unsigned)((ga >> 32) & 0x01FFFFFFu) | (2u << 30);  // [56:32]|type=2
    v8i g1;
    g1[0] = (int)(1u << 16);                      // wg_mask=0, data_size=2B
    g1[1] = (int)((tw & 0xFFFFu) << 16);          // tensor_dim0[15:0]
    g1[2] = (int)((tw >> 16) | ((th & 0xFFFFu) << 16));   // dim0[31:16],dim1[15:0]
    g1[3] = (int)((th >> 16) | ((tile_w & 0xFFFFu) << 16)); // dim1[31:16],tile0
    g1[4] = (int)(tile_h & 0xFFFFu);              // tile_dim1 ; tile_dim2=0
    g1[5] = (int)stride;                          // dim0_stride[31:0]
    g1[6] = 0;                                    // dim0_stride[47:32], dim1_s lo
    g1[7] = 0;
    v4i z4 = {0, 0, 0, 0};
    v8i z8 = {0, 0, 0, 0, 0, 0, 0, 0};
    // 6-arg toolchain signature: (g0, g1, g2, g3, g4, cpol)
    __builtin_amdgcn_tensor_load_to_lds(g0, g1, z4, z4, z8, 0);
}
#define LDSOFF(p) ((unsigned)(unsigned long long)(p))
#endif

// -------------------------- sizes / workspace layout -----------------------
#define BATCH   2
#define SEQ     2048
#define NTOK    (BATCH * SEQ)          // 4096
#define DIMC    1024
#define HEADS   16
#define HDIM    64

#define OFF_QKV   ((size_t)0)                               // 4096*3072 f32
#define OFF_XBF   (OFF_QKV  + (size_t)NTOK * 3072 * 4)      // 4096*1024 bf16
#define OFF_WQKV  (OFF_XBF  + (size_t)NTOK * DIMC * 2)      // (3072,1024) bf16 ^T
#define OFF_WOUT  (OFF_WQKV + (size_t)DIMC * 3072 * 2)      // (1024,1024) bf16 ^T
#define OFF_QP    (OFF_WOUT + (size_t)DIMC * DIMC * 2)      // (b,h,n,d) bf16
#define OFF_KP    (OFF_QP   + (size_t)NTOK * DIMC * 2)
#define OFF_VT    (OFF_KP   + (size_t)NTOK * DIMC * 2)      // (b,h,d,n) bf16
#define OFF_ATTN  (OFF_VT   + (size_t)NTOK * DIMC * 2)      // (tok, 1024) bf16

// ----------------------------- fp32 -> bf16 --------------------------------
__global__ __launch_bounds__(256)
void cvt_bf16_kernel(const float* __restrict__ in, u16* __restrict__ out, int n4) {
    int i = blockIdx.x * 256 + threadIdx.x;
    if (i < n4) {
        float4 v = *(const float4*)(in + (size_t)i * 4);
        u16* o = out + (size_t)i * 4;
        o[0] = f2bf(v.x); o[1] = f2bf(v.y); o[2] = f2bf(v.z); o[3] = f2bf(v.w);
    }
}

// fp32 (K,N) row-major -> bf16 (N,K) row-major (weights transposed once)
__global__ __launch_bounds__(256)
void cvt_bf16_transpose_kernel(const float* __restrict__ in, u16* __restrict__ out,
                               int K, int N) {
    int n  = blockIdx.x * 32 + (threadIdx.x & 31);
    int k0 = blockIdx.y * 32 + (threadIdx.x >> 5) * 4;
    #pragma unroll
    for (int i = 0; i < 4; ++i)
        out[(size_t)n * K + k0 + i] = f2bf(in[(size_t)(k0 + i) * N + n]);
}

// ------------------------ tiled bf16 WMMA GEMM -----------------------------
// C[M,N](f32) = A[M,K](bf16,row-major) * Bt[N,K](bf16,row-major)^T + bias[N]
// 64x64 workgroup tile, 4 waves; TDM double-buffered LDS staging.
#define TM 64
#define TN 64
#define TK 32

__global__ __launch_bounds__(128)
void gemm_bf16_kernel(const u16* __restrict__ A, const u16* __restrict__ Bt,
                      const float* __restrict__ bias, float* __restrict__ C,
                      int M, int N, int K) {
    __shared__ __align__(16) u16 Asl[2][TM][TK];   // (m, k)
    __shared__ __align__(16) u16 Bsl[2][TN][TK];   // (n, k)

    const int tid  = threadIdx.x;
    const int wave = tid >> 5;
    const int lane = tid & 31;
    const int lhi  = lane >> 4, llo = lane & 15;
    const int mtile = blockIdx.y * TM;
    const int ntile = blockIdx.x * TN;
    const int m0 = wave * 16;

    v8f acc[4] = {};

#if USE_TDM
    const int T = K / TK;
    if (wave == 0) {
        tdm_load_2d(A  + (size_t)mtile * K, LDSOFF(&Asl[0][0][0]), K, M, K, TK, TM);
        tdm_load_2d(Bt + (size_t)ntile * K, LDSOFF(&Bsl[0][0][0]), K, N, K, TK, TN);
    }
    for (int t = 0; t < T; ++t) {
        const int cur = t & 1;
        if (wave == 0) {
            if (t + 1 < T) {
                tdm_load_2d(A  + (size_t)mtile * K + (t + 1) * TK,
                            LDSOFF(&Asl[cur ^ 1][0][0]), K, M, K, TK, TM);
                tdm_load_2d(Bt + (size_t)ntile * K + (t + 1) * TK,
                            LDSOFF(&Bsl[cur ^ 1][0][0]), K, N, K, TK, TN);
                __builtin_amdgcn_s_wait_tensorcnt(2);   // prev pair done (in-order)
            } else {
                __builtin_amdgcn_s_wait_tensorcnt(0);
            }
        }
        __syncthreads();

        Frag16 af;
        af.u4[0] = *(const uint4*)&Asl[cur][m0 + llo][lhi * 8];
        af.u4[1] = *(const uint4*)&Asl[cur][m0 + llo][lhi * 8 + 16];
        #pragma unroll
        for (int g = 0; g < 4; ++g) {
            Frag16 bfr;
            bfr.u4[0] = *(const uint4*)&Bsl[cur][g * 16 + llo][lhi * 8];
            bfr.u4[1] = *(const uint4*)&Bsl[cur][g * 16 + llo][lhi * 8 + 16];
            acc[g] = __builtin_amdgcn_wmma_f32_16x16x32_bf16(
                false, af.v, false, bfr.v, (short)0, acc[g], false, false);
        }
        __syncthreads();   // all reads done before TDM t+2 overwrites this buffer
    }
#else
    for (int kk = 0; kk < K; kk += TK) {
        // stage through registers, then barrier+store+barrier (both row-major)
        uint4 ar[2], br[2];
        #pragma unroll
        for (int i = 0; i < 2; ++i) {
            int idx8 = tid * 2 + i, row = idx8 >> 2, chunk = idx8 & 3;
            ar[i] = *(const uint4*)(A  + (size_t)(mtile + row) * K + kk + chunk * 8);
            br[i] = *(const uint4*)(Bt + (size_t)(ntile + row) * K + kk + chunk * 8);
        }
        if (kk + TK < K) {
            __builtin_prefetch(A  + (size_t)(mtile + (tid >> 1)) * K + kk + TK, 0, 3);
            __builtin_prefetch(Bt + (size_t)(ntile + (tid >> 1)) * K + kk + TK, 0, 3);
        }
        __syncthreads();
        #pragma unroll
        for (int i = 0; i < 2; ++i) {
            int idx8 = tid * 2 + i, row = idx8 >> 2, chunk = idx8 & 3;
            *(uint4*)&Asl[0][row][chunk * 8] = ar[i];
            *(uint4*)&Bsl[0][row][chunk * 8] = br[i];
        }
        __syncthreads();

        Frag16 af;
        af.u4[0] = *(const uint4*)&Asl[0][m0 + llo][lhi * 8];
        af.u4[1] = *(const uint4*)&Asl[0][m0 + llo][lhi * 8 + 16];
        #pragma unroll
        for (int g = 0; g < 4; ++g) {
            Frag16 bfr;
            bfr.u4[0] = *(const uint4*)&Bsl[0][g * 16 + llo][lhi * 8];
            bfr.u4[1] = *(const uint4*)&Bsl[0][g * 16 + llo][lhi * 8 + 16];
            acc[g] = __builtin_amdgcn_wmma_f32_16x16x32_bf16(
                false, af.v, false, bfr.v, (short)0, acc[g], false, false);
        }
    }
#endif

    // C/D layout: VGPR r holds row m0+r+8*lhi, col ntile+g*16+llo
    #pragma unroll
    for (int g = 0; g < 4; ++g) {
        int col = ntile + g * 16 + llo;
        float bv = bias[col];
        #pragma unroll
        for (int r = 0; r < 8; ++r) {
            int row = mtile + m0 + r + 8 * lhi;
            C[(size_t)row * N + col] = acc[g][r] + bv;
        }
    }
}

// -------------------- RMSNorm + RoPE + bf16 pack ---------------------------
__global__ __launch_bounds__(256)
void qknorm_rope_pack_kernel(const float* __restrict__ qkv,
                             const float* __restrict__ q_scale,
                             const float* __restrict__ k_scale,
                             u16* __restrict__ qp, u16* __restrict__ kp,
                             u16* __restrict__ vt) {
    const int t = blockIdx.x;               // token
    const int b = t >> 11, n = t & (SEQ - 1);
    const int tid = threadIdx.x;
    const int c0 = tid * 4;

    const float* qrow = qkv + (size_t)t * 3072;
    float4 q4 = *(const float4*)(qrow + c0);
    float4 k4 = *(const float4*)(qrow + DIMC + c0);
    float4 v4 = *(const float4*)(qrow + 2 * DIMC + c0);

    float sq = q4.x*q4.x + q4.y*q4.y + q4.z*q4.z + q4.w*q4.w;
    float sk = k4.x*k4.x + k4.y*k4.y + k4.z*k4.z + k4.w*k4.w;
    #pragma unroll
    for (int off = 16; off; off >>= 1) {
        sq += __shfl_xor(sq, off, 32);
        sk += __shfl_xor(sk, off, 32);
    }
    __shared__ float redq[8], redk[8], bc[2];
    if ((tid & 31) == 0) { redq[tid >> 5] = sq; redk[tid >> 5] = sk; }
    __syncthreads();
    if (tid == 0) {
        float aq = 0.f, ak = 0.f;
        #pragma unroll
        for (int i = 0; i < 8; ++i) { aq += redq[i]; ak += redk[i]; }
        bc[0] = rsqrtf(aq * (1.0f / DIMC) + 1e-6f);
        bc[1] = rsqrtf(ak * (1.0f / DIMC) + 1e-6f);
    }
    __syncthreads();
    const float rq = bc[0], rk = bc[1];

    float4 qs4 = *(const float4*)(q_scale + c0);
    float4 ks4 = *(const float4*)(k_scale + c0);
    float qv[4] = { q4.x*rq*qs4.x, q4.y*rq*qs4.y, q4.z*rq*qs4.z, q4.w*rq*qs4.w };
    float kv[4] = { k4.x*rk*ks4.x, k4.y*rk*ks4.y, k4.z*rk*ks4.z, k4.w*rk*ks4.w };
    float vv[4] = { v4.x, v4.y, v4.z, v4.w };

    if (n > 0) {                            // RoPE, pos = n-1
        float pos = (float)(n - 1);
        #pragma unroll
        for (int p = 0; p < 2; ++p) {
            int d_even = (c0 + 2 * p) & (HDIM - 1);
            float invf = __powf(10000.0f, -(float)d_even * (1.0f / HDIM));
            float s, c;
            __sincosf(pos * invf, &s, &c);
            float q0 = qv[2*p], q1 = qv[2*p+1];
            qv[2*p]   = q0 * c - q1 * s;
            qv[2*p+1] = q1 * c + q0 * s;
            float k0 = kv[2*p], k1 = kv[2*p+1];
            kv[2*p]   = k0 * c - k1 * s;
            kv[2*p+1] = k1 * c + k0 * s;
        }
    }

    const int h = c0 >> 6, dl = c0 & (HDIM - 1);
    u16* qo = qp + (((size_t)(b * HEADS + h) * SEQ + n) * HDIM + dl);
    u16* ko = kp + (((size_t)(b * HEADS + h) * SEQ + n) * HDIM + dl);
    #pragma unroll
    for (int j = 0; j < 4; ++j) { qo[j] = f2bf(qv[j]); ko[j] = f2bf(kv[j]); }
    #pragma unroll
    for (int j = 0; j < 4; ++j)             // V transposed: (b,h,d,n)
        vt[((size_t)(b * HEADS + h) * HDIM + dl + j) * SEQ + n] = f2bf(vv[j]);
}

// --------------------------- flash attention -------------------------------
// grid: (SEQ/64, B*H); 128 threads = 4 waves, 1 wave per 16-row Q block.
// TDM stages the (shared) 32-key K and V^T tiles in double-buffered LDS.
__global__ __launch_bounds__(128)
void attn_kernel(const u16* __restrict__ qp, const u16* __restrict__ kp,
                 const u16* __restrict__ vt, u16* __restrict__ attn) {
    const int bh   = blockIdx.y;
    const int wave = threadIdx.x >> 5;
    const int lane = threadIdx.x & 31;
    const int lhi  = lane >> 4, llo = lane & 15;
    const int qrow0 = blockIdx.x * 64 + wave * 16;

    const u16* qb = qp + (size_t)bh * SEQ * HDIM;
    const u16* kb = kp + (size_t)bh * SEQ * HDIM;
    const u16* vb = vt + (size_t)bh * HDIM * SEQ;

    __shared__ __align__(16) u16 Pl[4][16][32];       // per-wave P tile
#if USE_TDM
    __shared__ __align__(16) u16 Ksl[2][32][64];      // (key, chan)
    __shared__ __align__(16) u16 Vsl[2][64][32];      // (d, key)
#endif

    // Q fragments, channel halves 0-31 / 32-63
    Frag16 qa[2];
    {
        const u16* qr = qb + (size_t)(qrow0 + llo) * HDIM + lhi * 8;
        #pragma unroll
        for (int hf = 0; hf < 2; ++hf) {
            qa[hf].u4[0] = *(const uint4*)(qr + hf * 32);
            qa[hf].u4[1] = *(const uint4*)(qr + hf * 32 + 16);
        }
    }

    v8f acc[4] = {};
    float mrow[8], lrow[8];
    #pragma unroll
    for (int r = 0; r < 8; ++r) { mrow[r] = -1e30f; lrow[r] = 0.0f; }
    const float scale = 0.125f;   // 1/sqrt(64)

#if USE_TDM
    if (wave == 0) {
        tdm_load_2d(kb, LDSOFF(&Ksl[0][0][0]), HDIM, SEQ, HDIM, HDIM, 32);
        tdm_load_2d(vb, LDSOFF(&Vsl[0][0][0]), SEQ, HDIM, SEQ, 32, HDIM);
    }
#endif
    for (int t = 0; t < SEQ / 32; ++t) {
        const int kt = t * 32;
        const int cur = t & 1;
#if USE_TDM
        if (wave == 0) {
            if (kt + 32 < SEQ) {
                tdm_load_2d(kb + (size_t)(kt + 32) * HDIM,
                            LDSOFF(&Ksl[cur ^ 1][0][0]), HDIM, SEQ, HDIM, HDIM, 32);
                tdm_load_2d(vb + (kt + 32),
                            LDSOFF(&Vsl[cur ^ 1][0][0]), SEQ, HDIM, SEQ, 32, HDIM);
                __builtin_amdgcn_s_wait_tensorcnt(2);
            } else {
                __builtin_amdgcn_s_wait_tensorcnt(0);
            }
        }
        __syncthreads();
#endif
        // ---- S(16x32) = Q(16x64) K^T : 4 WMMAs
        v8f s[2] = {};
        #pragma unroll
        for (int nh = 0; nh < 2; ++nh) {
            #pragma unroll
            for (int ks = 0; ks < 2; ++ks) {
                Frag16 bfr;
#if USE_TDM
                const u16* kr = &Ksl[cur][nh * 16 + llo][ks * 32 + lhi * 8];
#else
                const u16* kr = kb + (size_t)(kt + nh * 16 + llo) * HDIM
                                   + ks * 32 + lhi * 8;
#endif
                bfr.u4[0] = *(const uint4*)(kr);
                bfr.u4[1] = *(const uint4*)(kr + 16);
                s[nh] = __builtin_amdgcn_wmma_f32_16x16x32_bf16(
                    false, qa[ks].v, false, bfr.v, (short)0, s[nh], false, false);
            }
        }

        // ---- online softmax (rows spread across 16-lane halves)
        float corr[8];
        #pragma unroll
        for (int r = 0; r < 8; ++r) {
            float v0 = s[0][r] * scale, v1 = s[1][r] * scale;
            float mx = fmaxf(v0, v1);
            #pragma unroll
            for (int off = 8; off; off >>= 1) mx = fmaxf(mx, __shfl_xor(mx, off, 16));
            float mnew = fmaxf(mrow[r], mx);
            float c  = __expf(mrow[r] - mnew);
            float p0 = __expf(v0 - mnew), p1 = __expf(v1 - mnew);
            float rs = p0 + p1;
            #pragma unroll
            for (int off = 8; off; off >>= 1) rs += __shfl_xor(rs, off, 16);
            lrow[r] = lrow[r] * c + rs;
            mrow[r] = mnew;
            corr[r] = c;
            Pl[wave][r + 8 * lhi][llo]      = f2bf(p0);   // C-layout -> LDS
            Pl[wave][r + 8 * lhi][llo + 16] = f2bf(p1);
        }
        #pragma unroll
        for (int g = 0; g < 4; ++g)
            #pragma unroll
            for (int r = 0; r < 8; ++r) acc[g][r] *= corr[r];

        // ---- re-gather P as A fragment (in-wave LDS, no barrier needed)
        Frag16 pa;
        pa.u4[0] = *(const uint4*)&Pl[wave][llo][lhi * 8];
        pa.u4[1] = *(const uint4*)&Pl[wave][llo][lhi * 8 + 16];

        // ---- O(16x64) += P(16x32) V(32x64) : 4 WMMAs (V in (d,key) layout)
        #pragma unroll
        for (int g = 0; g < 4; ++g) {
            Frag16 vf;
#if USE_TDM
            const u16* vr = &Vsl[cur][g * 16 + llo][lhi * 8];
#else
            const u16* vr = vb + (size_t)(g * 16 + llo) * SEQ + kt + lhi * 8;
#endif
            vf.u4[0] = *(const uint4*)(vr);
            vf.u4[1] = *(const uint4*)(vr + 16);
            acc[g] = __builtin_amdgcn_wmma_f32_16x16x32_bf16(
                false, pa.v, false, vf.v, (short)0, acc[g], false, false);
        }
#if USE_TDM
        __syncthreads();   // reads done before TDM t+2 overwrites this buffer
#endif
    }

    // ---- normalize and pack to (token, h*64+d) bf16 for the output GEMM
    const int b = bh >> 4, h = bh & 15;
    #pragma unroll
    for (int r = 0; r < 8; ++r) {
        float inv = lrow[r] > 0.0f ? 1.0f / lrow[r] : 0.0f;
        size_t tok = (size_t)b * SEQ + qrow0 + r + 8 * lhi;
        #pragma unroll
        for (int g = 0; g < 4; ++g)
            attn[tok * DIMC + h * HDIM + g * 16 + llo] = f2bf(acc[g][r] * inv);
    }
}

// ------------------------------- launcher ----------------------------------
extern "C" void kernel_launch(void* const* d_in, const int* in_sizes, int n_in,
                              void* d_out, int out_size, void* d_ws, size_t ws_size,
                              hipStream_t stream) {
    const float* x       = (const float*)d_in[0];
    const float* Wqkv    = (const float*)d_in[1];
    const float* bqkv    = (const float*)d_in[2];
    const float* q_scale = (const float*)d_in[3];
    const float* k_scale = (const float*)d_in[4];
    const float* Wout    = (const float*)d_in[5];
    const float* bout    = (const float*)d_in[6];
    float* out = (float*)d_out;

    char* ws = (char*)d_ws;
    float* qkv   = (float*)(ws + OFF_QKV);
    u16*   xbf   = (u16*)(ws + OFF_XBF);
    u16*   wqkvT = (u16*)(ws + OFF_WQKV);   // (3072,1024)
    u16*   woutT = (u16*)(ws + OFF_WOUT);   // (1024,1024)
    u16*   qp    = (u16*)(ws + OFF_QP);
    u16*   kp    = (u16*)(ws + OFF_KP);
    u16*   vtb   = (u16*)(ws + OFF_VT);
    u16*   attnb = (u16*)(ws + OFF_ATTN);

    // 1) operand conversion: x -> bf16 ; weights -> bf16 AND transposed
    cvt_bf16_kernel<<<(NTOK * DIMC / 4 + 255) / 256, 256, 0, stream>>>(
        x, xbf, NTOK * DIMC / 4);
    cvt_bf16_transpose_kernel<<<dim3(3072 / 32, DIMC / 32), 256, 0, stream>>>(
        Wqkv, wqkvT, DIMC, 3072);
    cvt_bf16_transpose_kernel<<<dim3(DIMC / 32, DIMC / 32), 256, 0, stream>>>(
        Wout, woutT, DIMC, DIMC);

    // 2) QKV GEMM: (4096x1024)(1024x3072) + bqkv -> qkv f32
    gemm_bf16_kernel<<<dim3(3072 / TN, NTOK / TM), 128, 0, stream>>>(
        xbf, wqkvT, bqkv, qkv, NTOK, 3072, DIMC);

    // 3) RMSNorm + RoPE + pack (Q,K as (b,h,n,d); V as (b,h,d,n))
    qknorm_rope_pack_kernel<<<NTOK, 256, 0, stream>>>(qkv, q_scale, k_scale, qp, kp, vtb);

    // 4) flash attention -> attn bf16 (token-major rows of 1024)
    attn_kernel<<<dim3(SEQ / 64, BATCH * HEADS), 128, 0, stream>>>(qp, kp, vtb, attnb);

    // 5) output GEMM: (4096x1024)(1024x1024) + bout -> d_out f32
    gemm_bf16_kernel<<<dim3(DIMC / TN, NTOK / TM), 128, 0, stream>>>(
        attnb, woutT, bout, out, NTOK, DIMC, DIMC);
}
